// MultiheadAttn_62895501082904
// MI455X (gfx1250) — matrix operand order repaired and checked by reading, not compiled, not run on hardware
//
#include <hip/hip_runtime.h>

// ---------------------------------------------------------------------------
// Retention-style multihead attention for MI455X (gfx1250, wave32, WMMA bf16)
// ---------------------------------------------------------------------------
#define D_MODEL 1024
#define HEADS   16
#define DK      64
#define SEQ     4096
#define BATCH   4
#define ROWS    (BATCH * SEQ)          // 16384

typedef __attribute__((ext_vector_type(16))) __bf16 v16bf;
typedef __attribute__((ext_vector_type(8)))  __bf16 v8bf;
typedef __attribute__((ext_vector_type(8)))  float  v8f;

// Load an A fragment (16x32 bf16, M = lane%16) from a row pointer already
// offset to the K-base. Contiguous layout: lanes 0-15 hold K[0..15];
// lanes 16-31 hold K[16..31].
__device__ __forceinline__ v16bf load_a_frag(const __bf16* p, int half) {
  v8bf lo = *(const v8bf*)(p + half * 16);
  v8bf hi = *(const v8bf*)(p + half * 16 + 8);
  v16bf a;
#pragma unroll
  for (int i = 0; i < 8; ++i) { a[i] = lo[i]; a[i + 8] = hi[i]; }
  return a;
}

#define WMMA_BF16(a, b, c) \
  __builtin_amdgcn_wmma_f32_16x16x32_bf16(false, (a), false, (b), (short)0, (c), false, false)

// ---------------------------------------------------------------------------
// f32 -> bf16 conversion
// ---------------------------------------------------------------------------
__global__ void cvt_bf16_kernel(const float* __restrict__ in,
                                __bf16* __restrict__ out, int n) {
  int i = blockIdx.x * blockDim.x + threadIdx.x;
  if (i < n) out[i] = (__bf16)in[i];
}

// ---------------------------------------------------------------------------
// C[M,N] = A[M,K] @ W[N,K]^T   (bf16 in, f32 out)
// Wave tile 32(M) x 64(N) (2 A frags x 4 B frags = 8 WMMA / K-step),
// 8 waves (4 M x 2 N) -> 128 x 128 per block.
// ---------------------------------------------------------------------------
__global__ __launch_bounds__(256)
void gemm_bf16_kernel(const __bf16* __restrict__ A, const __bf16* __restrict__ W,
                      float* __restrict__ C, int M, int N, int K) {
  const int lane = threadIdx.x & 31;
  const int wid  = threadIdx.x >> 5;
  const int half = lane >> 4;
  const int col  = lane & 15;
  const int m0 = blockIdx.y * 128 + (wid & 3) * 32;
  const int n0 = blockIdx.x * 128 + (wid >> 2) * 64;

  const __bf16* arow0 = A + (size_t)(m0 + col) * K;
  const __bf16* arow1 = A + (size_t)(m0 + 16 + col) * K;
  const __bf16* brow  = W + (size_t)(n0 + col) * K + half * 16;

  v8f acc0_0 = {}, acc0_1 = {}, acc0_2 = {}, acc0_3 = {};
  v8f acc1_0 = {}, acc1_1 = {}, acc1_2 = {}, acc1_3 = {};

  for (int kb = 0; kb < K; kb += 32) {
    // prefetch next K-step of the weight stream (global_prefetch_b8)
    if (kb + 32 < K) {
      __builtin_prefetch(brow + kb + 32, 0, 3);
      __builtin_prefetch(brow + (size_t)32 * K + kb + 32, 0, 3);
    }
    v16bf a0 = load_a_frag(arow0 + kb, half);
    v16bf a1 = load_a_frag(arow1 + kb, half);
    v16bf b0 = *(const v16bf*)(brow + kb);
    v16bf b1 = *(const v16bf*)(brow + (size_t)16 * K + kb);
    v16bf b2 = *(const v16bf*)(brow + (size_t)32 * K + kb);
    v16bf b3 = *(const v16bf*)(brow + (size_t)48 * K + kb);
    acc0_0 = WMMA_BF16(a0, b0, acc0_0);
    acc1_0 = WMMA_BF16(a1, b0, acc1_0);
    acc0_1 = WMMA_BF16(a0, b1, acc0_1);
    acc1_1 = WMMA_BF16(a1, b1, acc1_1);
    acc0_2 = WMMA_BF16(a0, b2, acc0_2);
    acc1_2 = WMMA_BF16(a1, b2, acc1_2);
    acc0_3 = WMMA_BF16(a0, b3, acc0_3);
    acc1_3 = WMMA_BF16(a1, b3, acc1_3);
  }
#pragma unroll
  for (int r = 0; r < 8; ++r) {
    float* out0 = C + (size_t)(m0 + r + half * 8) * N + n0 + col;
    out0[0]  = acc0_0[r];
    out0[16] = acc0_1[r];
    out0[32] = acc0_2[r];
    out0[48] = acc0_3[r];
    float* out1 = C + (size_t)(m0 + 16 + r + half * 8) * N + n0 + col;
    out1[0]  = acc1_0[r];
    out1[16] = acc1_1[r];
    out1[32] = acc1_2[r];
    out1[48] = acc1_3[r];
  }
}

// ---------------------------------------------------------------------------
// xPos rotary on query/key; repack q (row-major bf16) and k/v transposed to
// [b,h,d,s] so downstream einsum B operands are K-contiguous.
// One thread per (b,s,pair p), p in [0,512).
// ---------------------------------------------------------------------------
__global__ void xpos_repack_kernel(const float* __restrict__ qf,
                                   const float* __restrict__ kf,
                                   const float* __restrict__ vf,
                                   __bf16* __restrict__ q_bf,
                                   __bf16* __restrict__ khT,
                                   __bf16* __restrict__ vhT) {
  size_t idx = (size_t)blockIdx.x * blockDim.x + threadIdx.x;
  int p = (int)(idx & 511);
  size_t row = idx >> 9;                  // b*SEQ + s
  int s = (int)(row & (SEQ - 1));
  int b = (int)(row >> 12);
  float pos = (float)s;

  float sv    = (2.0f * p + 0.4f * D_MODEL) / (1.4f * D_MODEL);
  float scale = __powf(sv, pos * (1.0f / 512.0f));
  float ang   = pos * __powf(10000.0f, -(float)p * (1.0f / 512.0f));
  float sn, cs;
  __sincosf(ang, &sn, &cs);
  sn *= scale; cs *= scale;

  size_t off = row * D_MODEL + 2 * p;
  float qe = qf[off], qo = qf[off + 1];
  float ke = kf[off], ko = kf[off + 1];
  q_bf[off]     = (__bf16)(qe * cs - qo * sn);
  q_bf[off + 1] = (__bf16)(qo * cs + qe * sn);

  int j = 2 * p;
  int h = j >> 6, d = j & 63;
  size_t t = (((size_t)b * HEADS + h) * DK + d) * SEQ + s;
  khT[t]       = (__bf16)(ke * cs - ko * sn);
  khT[t + SEQ] = (__bf16)(ko * cs + ke * sn);
  vhT[t]       = (__bf16)vf[off];
  vhT[t + SEQ] = (__bf16)vf[off + 1];
}

// ---------------------------------------------------------------------------
// kp/vp einsum: out[e,d] = sum_s wx[e,s] * xhT[d,s]  (+ wx_b[e])
// kpT stored [e,d] (scores B operand), vp stored [d,e] (y B operand).
// One wave: 32(e) x 64(d) tile, K = 4096.  256 waves total.
// ---------------------------------------------------------------------------
__global__ __launch_bounds__(256)
void px_einsum_kernel(const __bf16* __restrict__ wx, const float* __restrict__ wxb,
                      const __bf16* __restrict__ khT, const __bf16* __restrict__ vhT,
                      __bf16* __restrict__ kpT, __bf16* __restrict__ vp) {
  const int lane = threadIdx.x & 31;
  const int wid  = threadIdx.x >> 5;
  const int half = lane >> 4;
  const int col  = lane & 15;
  int id  = blockIdx.x * 8 + wid;       // 0..255
  int bh  = id >> 2;                    // 0..63
  int src = (id >> 1) & 1;              // 0: key path, 1: value path
  int e0  = (id & 1) * 32;

  const __bf16* B = (src == 0 ? khT : vhT) + (size_t)bh * DK * SEQ;
  const __bf16* arow0 = wx + (size_t)(e0 + col) * SEQ;
  const __bf16* arow1 = wx + (size_t)(e0 + 16 + col) * SEQ;
  const __bf16* brow  = B + (size_t)col * SEQ + half * 16;

  v8f acc0_0 = {}, acc0_1 = {}, acc0_2 = {}, acc0_3 = {};
  v8f acc1_0 = {}, acc1_1 = {}, acc1_2 = {}, acc1_3 = {};
  for (int kb = 0; kb < SEQ; kb += 32) {
    v16bf a0 = load_a_frag(arow0 + kb, half);
    v16bf a1 = load_a_frag(arow1 + kb, half);
    v16bf b0 = *(const v16bf*)(brow + kb);
    v16bf b1 = *(const v16bf*)(brow + (size_t)16 * SEQ + kb);
    v16bf b2 = *(const v16bf*)(brow + (size_t)32 * SEQ + kb);
    v16bf b3 = *(const v16bf*)(brow + (size_t)48 * SEQ + kb);
    acc0_0 = WMMA_BF16(a0, b0, acc0_0);
    acc1_0 = WMMA_BF16(a1, b0, acc1_0);
    acc0_1 = WMMA_BF16(a0, b1, acc0_1);
    acc1_1 = WMMA_BF16(a1, b1, acc1_1);
    acc0_2 = WMMA_BF16(a0, b2, acc0_2);
    acc1_2 = WMMA_BF16(a1, b2, acc1_2);
    acc0_3 = WMMA_BF16(a0, b3, acc0_3);
    acc1_3 = WMMA_BF16(a1, b3, acc1_3);
  }

  __bf16* okp = kpT + (size_t)bh * DK * DK;
  __bf16* ovp = vp  + (size_t)bh * DK * DK;
#pragma unroll
  for (int part = 0; part < 2; ++part) {
    v8f* c0 = part ? &acc1_0 : &acc0_0;
    v8f* c1 = part ? &acc1_1 : &acc0_1;
    v8f* c2 = part ? &acc1_2 : &acc0_2;
    v8f* c3 = part ? &acc1_3 : &acc0_3;
#pragma unroll
    for (int r = 0; r < 8; ++r) {
      int e = e0 + part * 16 + r + half * 8;
      float bias = wxb[e];
      float v0 = (*c0)[r] + bias, v1 = (*c1)[r] + bias;
      float v2 = (*c2)[r] + bias, v3 = (*c3)[r] + bias;
      if (src == 0) {        // kpT[e, d]
        __bf16* o = okp + (size_t)e * DK + col;
        o[0] = (__bf16)v0; o[16] = (__bf16)v1; o[32] = (__bf16)v2; o[48] = (__bf16)v3;
      } else {               // vp[d, e]  (transposed store)
        ovp[(size_t)(col +  0) * DK + e] = (__bf16)v0;
        ovp[(size_t)(col + 16) * DK + e] = (__bf16)v1;
        ovp[(size_t)(col + 32) * DK + e] = (__bf16)v2;
        ovp[(size_t)(col + 48) * DK + e] = (__bf16)v3;
      }
    }
  }
}

// ---------------------------------------------------------------------------
// Attention: per wave one (b,h,16-row s-tile).
// scores = qh @ kpT (via WMMA) * decay, softmax over 64, y = attn @ vp^T.
// Softmax staged through wave-private LDS.
// ---------------------------------------------------------------------------
__global__ __launch_bounds__(256)
void attention_kernel(const __bf16* __restrict__ qbf, const __bf16* __restrict__ kpT,
                      const __bf16* __restrict__ vp, float* __restrict__ y) {
  __shared__ __align__(32) float  sc[8][16 * 64];
  __shared__ __align__(32) __bf16 at[8][16 * 64];

  const int lane = threadIdx.x & 31;
  const int wid  = threadIdx.x >> 5;
  const int half = lane >> 4;
  const int col  = lane & 15;
  int id    = blockIdx.x * 8 + wid;     // 0..16383
  int stile = id & 255;
  int bh    = id >> 8;
  int b = bh >> 4, h = bh & 15;
  int s0 = stile * 16;

  // ---- scores = qh @ kp (M=16 rows of s, N=64 cols of e, K=64 of d) ----
  const __bf16* arow = qbf + ((size_t)b * SEQ + s0 + col) * D_MODEL + h * DK;
  const __bf16* kpb  = kpT + (size_t)bh * DK * DK + (size_t)col * DK + half * 16;
  v8f c0 = {}, c1 = {}, c2 = {}, c3 = {};
#pragma unroll
  for (int kb = 0; kb < DK; kb += 32) {
    v16bf a  = load_a_frag(arow + kb, half);
    v16bf b0 = *(const v16bf*)(kpb + kb);
    v16bf b1 = *(const v16bf*)(kpb + 16 * DK + kb);
    v16bf b2 = *(const v16bf*)(kpb + 32 * DK + kb);
    v16bf b3 = *(const v16bf*)(kpb + 48 * DK + kb);
    c0 = WMMA_BF16(a, b0, c0);
    c1 = WMMA_BF16(a, b1, c1);
    c2 = WMMA_BF16(a, b2, c2);
    c3 = WMMA_BF16(a, b3, c3);
  }

  // ---- retention decay Dt[h,s,e] = (e>=s) ? gamma^(e-s) : 0, into LDS ----
  float gamma = 1.0f - (1.0f / 32.0f) * __powf(1.0f / 16.0f, (float)h / 15.0f);
  float* stile_p = sc[wid];
#pragma unroll
  for (int r = 0; r < 8; ++r) {
    int m = r + half * 8;
    int sg = s0 + m;
    {
      int e = col;       int dd = e - sg;
      stile_p[m * 64 + e] = (dd >= 0) ? c0[r] * __powf(gamma, (float)dd) : 0.0f;
    }
    {
      int e = col + 16;  int dd = e - sg;
      stile_p[m * 64 + e] = (dd >= 0) ? c1[r] * __powf(gamma, (float)dd) : 0.0f;
    }
    {
      int e = col + 32;  int dd = e - sg;
      stile_p[m * 64 + e] = (dd >= 0) ? c2[r] * __powf(gamma, (float)dd) : 0.0f;
    }
    {
      int e = col + 48;  int dd = e - sg;
      stile_p[m * 64 + e] = (dd >= 0) ? c3[r] * __powf(gamma, (float)dd) : 0.0f;
    }
  }

  // ---- softmax over 64 cols: lane handles (row = col, half-row of 32) ----
  float* rowp = &stile_p[col * 64 + half * 32];
  float mx = -3.4e38f;
#pragma unroll
  for (int i = 0; i < 32; ++i) mx = fmaxf(mx, rowp[i]);
  mx = fmaxf(mx, __shfl_xor(mx, 16, 32));
  float sum = 0.0f;
#pragma unroll
  for (int i = 0; i < 32; ++i) { float ex = __expf(rowp[i] - mx); rowp[i] = ex; sum += ex; }
  sum += __shfl_xor(sum, 16, 32);
  float inv = 1.0f / sum;
  __bf16* atile = at[wid];
#pragma unroll
  for (int i = 0; i < 32; ++i)
    atile[col * 64 + half * 32 + i] = (__bf16)(rowp[i] * inv);

  // ---- y = attn @ vp^T (M=16, N=64 of f, K=64 of e) ----
  const __bf16* vpb = vp + (size_t)bh * DK * DK + (size_t)col * DK + half * 16;
  v8f d0 = {}, d1 = {}, d2 = {}, d3 = {};
#pragma unroll
  for (int kb = 0; kb < DK; kb += 32) {
    v16bf a  = load_a_frag(atile + col * 64 + kb, half);
    v16bf b0 = *(const v16bf*)(vpb + kb);
    v16bf b1 = *(const v16bf*)(vpb + 16 * DK + kb);
    v16bf b2 = *(const v16bf*)(vpb + 32 * DK + kb);
    v16bf b3 = *(const v16bf*)(vpb + 48 * DK + kb);
    d0 = WMMA_BF16(a, b0, d0);
    d1 = WMMA_BF16(a, b1, d1);
    d2 = WMMA_BF16(a, b2, d2);
    d3 = WMMA_BF16(a, b3, d3);
  }
#pragma unroll
  for (int r = 0; r < 8; ++r) {
    float* out = y + ((size_t)b * SEQ + s0 + r + half * 8) * D_MODEL + h * DK + col;
    out[0]  = d0[r];
    out[16] = d1[r];
    out[32] = d2[r];
    out[48] = d3[r];
  }
}

// ---------------------------------------------------------------------------
// GroupNorm(groups of 4 within each 64-chunk) + gate + SiLU -> bf16
// One thread per group of 4 contiguous elements.
// ---------------------------------------------------------------------------
__global__ void gn_gate_kernel(const float* __restrict__ y, const float* __restrict__ gatep,
                               const float* __restrict__ gnw, const float* __restrict__ gnb,
                               __bf16* __restrict__ out) {
  size_t g = (size_t)blockIdx.x * blockDim.x + threadIdx.x;
  size_t base = g * 4;
  int p = (int)(base & 63);
  float v0 = y[base], v1 = y[base + 1], v2 = y[base + 2], v3 = y[base + 3];
  float mu = (v0 + v1 + v2 + v3) * 0.25f;
  float a0 = v0 - mu, a1 = v1 - mu, a2 = v2 - mu, a3 = v3 - mu;
  float var = (a0 * a0 + a1 * a1 + a2 * a2 + a3 * a3) * 0.25f;
  float inv = rsqrtf(var + 1e-5f);
#pragma unroll
  for (int i = 0; i < 4; ++i) {
    float z = (y[base + i] - mu) * inv * gnw[p + i] + gnb[p + i];
    float x = gatep[base + i] * z;
    float s = x / (1.0f + __expf(-x));
    out[base + i] = (__bf16)s;
  }
}

// ---------------------------------------------------------------------------
// Launch
// ---------------------------------------------------------------------------
extern "C" void kernel_launch(void* const* d_in, const int* in_sizes, int n_in,
                              void* d_out, int out_size, void* d_ws, size_t ws_size,
                              hipStream_t stream) {
  const float* q    = (const float*)d_in[0];
  const float* k    = (const float*)d_in[1];
  const float* v    = (const float*)d_in[2];
  const float* wq   = (const float*)d_in[3];
  const float* wk   = (const float*)d_in[4];
  const float* wv   = (const float*)d_in[5];
  const float* wo   = (const float*)d_in[6];
  const float* wg   = (const float*)d_in[7];
  const float* wx_w = (const float*)d_in[8];
  const float* wx_b = (const float*)d_in[9];
  const float* gn_w = (const float*)d_in[10];
  const float* gn_b = (const float*)d_in[11];

  char* ws = (char*)d_ws;
  size_t off = 0;
  auto take = [&](size_t bytes) {
    char* p = ws + off;
    off += (bytes + 255) & ~(size_t)255;
    return p;
  };
  const size_t WB = (size_t)D_MODEL * D_MODEL * sizeof(__bf16);   // 2 MB
  const size_t AB = (size_t)ROWS * D_MODEL * sizeof(__bf16);      // 32 MB
  const size_t FB = (size_t)ROWS * D_MODEL * sizeof(float);       // 64 MB
  const size_t PB = (size_t)BATCH * HEADS * DK * DK * sizeof(__bf16);

  __bf16* wq_bf = (__bf16*)take(WB);
  __bf16* wk_bf = (__bf16*)take(WB);
  __bf16* wv_bf = (__bf16*)take(WB);
  __bf16* wg_bf = (__bf16*)take(WB);
  __bf16* wo_bf = (__bf16*)take(WB);
  __bf16* wx_bf = (__bf16*)take((size_t)DK * SEQ * sizeof(__bf16));
  __bf16* qin   = (__bf16*)take(AB);   // aliased later: q_bf (post-xpos)
  __bf16* kin   = (__bf16*)take(AB);   // aliased later: khT
  __bf16* vin   = (__bf16*)take(AB);   // aliased later: vhT
  float* query_f = (float*)take(FB);
  float* key_f   = (float*)take(FB);   // aliased later: silu_bf
  float* value_f = (float*)take(FB);   // aliased later: y_f
  float* gate_f  = (float*)take(FB);
  __bf16* kpT = (__bf16*)take(PB);
  __bf16* vp  = (__bf16*)take(PB);

  __bf16* q_bf    = qin;               // post-xpos query, same layout/size
  __bf16* khT     = kin;               // [b,h,d,s]
  __bf16* vhT     = vin;               // [b,h,d,s]
  float*  y_f     = value_f;           // attention output
  __bf16* silu_bf = (__bf16*)key_f;    // gated SiLU activation

  const int nAct = ROWS * D_MODEL;     // 16,777,216
  const int nW   = D_MODEL * D_MODEL;  // 1,048,576
  const int nWx  = DK * SEQ;           // 262,144

  // 1) f32 -> bf16 conversions
  cvt_bf16_kernel<<<nAct / 256, 256, 0, stream>>>(q, qin, nAct);
  cvt_bf16_kernel<<<nAct / 256, 256, 0, stream>>>(k, kin, nAct);
  cvt_bf16_kernel<<<nAct / 256, 256, 0, stream>>>(v, vin, nAct);
  cvt_bf16_kernel<<<nW / 256, 256, 0, stream>>>(wq, wq_bf, nW);
  cvt_bf16_kernel<<<nW / 256, 256, 0, stream>>>(wk, wk_bf, nW);
  cvt_bf16_kernel<<<nW / 256, 256, 0, stream>>>(wv, wv_bf, nW);
  cvt_bf16_kernel<<<nW / 256, 256, 0, stream>>>(wg, wg_bf, nW);
  cvt_bf16_kernel<<<nW / 256, 256, 0, stream>>>(wo, wo_bf, nW);
  cvt_bf16_kernel<<<nWx / 256, 256, 0, stream>>>(wx_w, wx_bf, nWx);

  // 2) Projections (WMMA bf16): 128x128 block tiles
  dim3 ggrid(D_MODEL / 128, ROWS / 128);
  gemm_bf16_kernel<<<ggrid, 256, 0, stream>>>(qin, wq_bf, query_f, ROWS, D_MODEL, D_MODEL);
  gemm_bf16_kernel<<<ggrid, 256, 0, stream>>>(kin, wk_bf, key_f,   ROWS, D_MODEL, D_MODEL);
  gemm_bf16_kernel<<<ggrid, 256, 0, stream>>>(vin, wv_bf, value_f, ROWS, D_MODEL, D_MODEL);
  gemm_bf16_kernel<<<ggrid, 256, 0, stream>>>(qin, wg_bf, gate_f,  ROWS, D_MODEL, D_MODEL);

  // 3) xPos + repack (overwrites qin/kin/vin, which are now dead)
  xpos_repack_kernel<<<(ROWS * 512) / 256, 256, 0, stream>>>(
      query_f, key_f, value_f, q_bf, khT, vhT);

  // 4) kp/vp einsums (WMMA, K = 4096): 256 waves -> 32 blocks
  px_einsum_kernel<<<32, 256, 0, stream>>>(wx_bf, wx_b, khT, vhT, kpT, vp);

  // 5) Attention + decay + softmax (WMMA + LDS)
  attention_kernel<<<(BATCH * HEADS * (SEQ / 16)) / 8, 256, 0, stream>>>(
      q_bf, kpT, vp, y_f);

  // 6) GroupNorm + gate + SiLU
  gn_gate_kernel<<<(nAct / 4) / 256, 256, 0, stream>>>(y_f, gate_f, gn_w, gn_b, silu_bf);

  // 7) Output projection -> d_out (f32)
  gemm_bf16_kernel<<<ggrid, 256, 0, stream>>>(silu_bf, wo_bf, (float*)d_out,
                                              ROWS, D_MODEL, D_MODEL);
}
